// CLIPEncoderLayer_88965952569976
// MI455X (gfx1250) — matrix operand order repaired and checked by reading, not compiled, not run on hardware
//
#include <hip/hip_runtime.h>
#include <hip/hip_bf16.h>
#include <stdint.h>

typedef __attribute__((ext_vector_type(16))) __bf16 v16bf;
typedef __attribute__((ext_vector_type(8)))  __bf16 v8bf;
typedef __attribute__((ext_vector_type(8)))  float  v8f;

union BV { v16bf v; v8bf h[2]; };

#if __has_builtin(__builtin_amdgcn_global_load_async_to_lds_b128)
#define HAVE_ASYNC_LDS 1
typedef int v4i_vs __attribute__((__vector_size__(16)));   // matches builtin param type
#endif

// ---------------------------------------------------------------- cast f32->bf16
__global__ __launch_bounds__(256) void cast_f32_bf16_k(const float* __restrict__ in,
                                                       __bf16* __restrict__ out, int n) {
  int i = blockIdx.x * 256 + threadIdx.x;
  if (i < n) out[i] = (__bf16)in[i];
}

// ---------------------------------------------------------------- LayerNorm (row = 1024) -> bf16
__global__ __launch_bounds__(256) void ln_bf16_k(const float* __restrict__ x,
                                                 const float* __restrict__ g,
                                                 const float* __restrict__ beta,
                                                 __bf16* __restrict__ out) {
  const int N = 1024;
  const int row = blockIdx.x;
  const float* xr = x + (size_t)row * N;
  float v[4];
  float s = 0.f, s2 = 0.f;
#pragma unroll
  for (int t = 0; t < 4; ++t) {
    v[t] = xr[threadIdx.x + t * 256];
    s += v[t]; s2 += v[t] * v[t];
  }
  __shared__ float rs[256], rs2[256];
  rs[threadIdx.x] = s; rs2[threadIdx.x] = s2;
  __syncthreads();
  for (int o = 128; o > 0; o >>= 1) {
    if ((int)threadIdx.x < o) {
      rs[threadIdx.x]  += rs[threadIdx.x + o];
      rs2[threadIdx.x] += rs2[threadIdx.x + o];
    }
    __syncthreads();
  }
  float mean = rs[0] * (1.f / N);
  float var  = rs2[0] * (1.f / N) - mean * mean;
  float rstd = rsqrtf(var + 1e-5f);
  __bf16* orow = out + (size_t)row * N;
#pragma unroll
  for (int t = 0; t < 4; ++t) {
    int i = threadIdx.x + t * 256;
    orow[i] = (__bf16)((v[t] - mean) * rstd * g[i] + beta[i]);
  }
}

// ---------------------------------------------------------------- GEMM: Y = A @ W^T + bias (+epilogue)
// A: [M,K] bf16 row-major.  W: [N,K] bf16 row-major.
// Block = 256 thr (8 waves): tile 128(M) x 64(N); wave tile 16x64 (4 WMMA accs).
// B slab (64 rows x 32 k) staged in LDS per k-step, double-buffered via async-to-LDS.
// MODE 0: outb = (acc+bias)*scale          MODE 1: head-transposed V write
// MODE 2: outf = acc+bias+res (fp32)       MODE 3: outb = quick_gelu(acc+bias)
template <int MODE>
__global__ __launch_bounds__(256) void gemm_bf16_k(const __bf16* __restrict__ A,
                                                   const __bf16* __restrict__ Wt,
                                                   const float* __restrict__ bias,
                                                   const float* __restrict__ res,
                                                   float* __restrict__ outf,
                                                   __bf16* __restrict__ outb,
                                                   int M, int N, int K, float scale) {
  __shared__ __align__(16) __bf16 bst[2][64 * 40];   // 40-half pitch: 16B aligned, bank-spread
  const int tid  = threadIdx.x;
  const int wave = tid >> 5;
  const int lane = tid & 31;
  const int r  = lane & 15;
  const int hi = lane >> 4;
  const int m0 = blockIdx.y * 128 + wave * 16;
  const int n0 = blockIdx.x * 64;

  // staging assignment: thread stages 16B: row tid/4 of the slab, 8-half chunk tid%4
  const int trow = tid >> 2;
  const int tc   = (tid & 3) * 8;
  const __bf16* gBrow = Wt + (size_t)(n0 + trow) * K + tc;
  __bf16* sBdst0 = &bst[0][0] + trow * 40 + tc;
  __bf16* sBdst1 = &bst[1][0] + trow * 40 + tc;

  const __bf16* arow = A + (size_t)(m0 + r) * K;
  v8f acc[4] = {};

  const int nsteps = K >> 5;

#ifdef HAVE_ASYNC_LDS
  __builtin_amdgcn_global_load_async_to_lds_b128((v4i_vs*)gBrow, (v4i_vs*)sBdst0, 0, 0);
#else
  v8bf breg = *(const v8bf*)gBrow;
#endif

  for (int s = 0; s < nsteps; ++s) {
    const int k0 = s << 5;
#ifdef HAVE_ASYNC_LDS
    if (s + 1 < nsteps) {
      __builtin_amdgcn_global_load_async_to_lds_b128(
          (v4i_vs*)(gBrow + k0 + 32),
          (v4i_vs*)(((s + 1) & 1) ? sBdst1 : sBdst0), 0, 0);
      asm volatile("s_wait_asynccnt 0x1" ::: "memory");
    } else {
      asm volatile("s_wait_asynccnt 0x0" ::: "memory");
    }
#else
    *(v8bf*)((s & 1) ? sBdst1 : sBdst0) = breg;
    if (s + 1 < nsteps) breg = *(const v8bf*)(gBrow + k0 + 32);
#endif
    __syncthreads();

    BV a;  // A-layout: lane=row; lanes0-15 hold k0..+7 & k0+16..23, lanes16-31 the +8 halves
    a.h[0] = *(const v8bf*)(arow + k0 + hi * 8);
    a.h[1] = *(const v8bf*)(arow + k0 + 16 + hi * 8);
    __builtin_prefetch(arow + k0 + 64, 0, 1);   // global_prefetch_b8, two k-steps ahead

    const __bf16* bb = &bst[s & 1][0];
#pragma unroll
    for (int j = 0; j < 4; ++j) {
      const __bf16* bp = bb + (j * 16 + r) * 40 + hi * 8;   // B-layout: lane=column
      BV b;
      b.h[0] = *(const v8bf*)bp;
      b.h[1] = *(const v8bf*)(bp + 16);
      acc[j] = __builtin_amdgcn_wmma_f32_16x16x32_bf16(false, a.v, false, b.v,
                                                       (short)0, acc[j], false, false);
    }
    __syncthreads();
  }

#pragma unroll
  for (int j = 0; j < 4; ++j) {
    const int n = n0 + j * 16 + r;
    const float bs = bias[n];
#pragma unroll
    for (int i = 0; i < 8; ++i) {
      const int m = m0 + i + hi * 8;               // C layout: VGPR i -> row i (+8 for hi half)
      const size_t idx = (size_t)m * N + n;
      float v = acc[j][i] + bs;
      if constexpr (MODE == 0) {
        outb[idx] = (__bf16)(v * scale);
      } else if constexpr (MODE == 1) {
        const int b_ = m >> 10, s_ = m & 1023, h_ = n >> 6, d_ = n & 63;
        outb[((size_t)((b_ * 16 + h_) * 64 + d_) << 10) + s_] = (__bf16)v;
      } else if constexpr (MODE == 2) {
        outf[idx] = v + res[idx];
      } else {
        outb[idx] = (__bf16)(v * (1.0f / (1.0f + __expf(-1.702f * v))));
      }
    }
  }
}

// ---------------------------------------------------------------- flash attention
// q,k: [B,S,D] bf16 (q pre-scaled by Dh^-0.5).  vt: [B*H][64][S] bf16.  out: [B,S,D] bf16.
// grid (S/128, B*H), block 256: one wave per 16 query rows, key blocks of 32.
// Row-sums computed with an extra WMMA against a ones matrix (no sum butterfly).
__global__ __launch_bounds__(256) void attn_k(const __bf16* __restrict__ q,
                                              const __bf16* __restrict__ kmat,
                                              const __bf16* __restrict__ vt,
                                              __bf16* __restrict__ out) {
  __shared__ __align__(16) __bf16 pl[8][16][40];   // per-wave P transpose buffer
  const int wave = threadIdx.x >> 5;
  const int lane = threadIdx.x & 31;
  const int r  = lane & 15;
  const int hi = lane >> 4;
  const int bh = blockIdx.y, b = bh >> 4, h = bh & 15;
  const int q0 = blockIdx.x * 128 + wave * 16;

  // Q tile (16x64) as two A-layout 16x32 operands, loaded once
  const __bf16* qrow = q + ((size_t)(b * 1024 + q0 + r) << 10) + h * 64;
  BV aq[2];
#pragma unroll
  for (int c = 0; c < 2; ++c) {
    aq[c].h[0] = *(const v8bf*)(qrow + c * 32 + hi * 8);
    aq[c].h[1] = *(const v8bf*)(qrow + c * 32 + 16 + hi * 8);
  }

  BV onesb;   // 32x16 ones B-operand for row sums
#pragma unroll
  for (int e = 0; e < 16; ++e) onesb.v[e] = (__bf16)1.0f;

  float mrow[8], lrow[8];
#pragma unroll
  for (int i = 0; i < 8; ++i) { mrow[i] = -__builtin_inff(); lrow[i] = 0.f; }
  v8f o[4] = {};

  const __bf16* kbase = kmat + ((size_t)(b * 1024) << 10) + h * 64;
  const __bf16* vbase = vt + ((size_t)bh << 16);   // bh * 64 * 1024

  for (int kb = 0; kb < 1024; kb += 32) {
    // S = Q @ K^T : 16 queries x 32 keys, two 16x16 tiles
    v8f s[2] = {};
#pragma unroll
    for (int j = 0; j < 2; ++j) {
#pragma unroll
      for (int c = 0; c < 2; ++c) {
        const __bf16* kp = kbase + ((size_t)(kb + j * 16 + r) << 10) + c * 32 + hi * 8;
        BV bk;
        bk.h[0] = *(const v8bf*)kp;
        bk.h[1] = *(const v8bf*)(kp + 16);
        s[j] = __builtin_amdgcn_wmma_f32_16x16x32_bf16(false, aq[c].v, false, bk.v,
                                                       (short)0, s[j], false, false);
      }
    }
    // online softmax (row i+hi*8 lives in acc element i across 16 lanes of the half)
#pragma unroll
    for (int i = 0; i < 8; ++i) {
      float mx = fmaxf(s[0][i], s[1][i]);
#pragma unroll
      for (int m = 8; m >= 1; m >>= 1) mx = fmaxf(mx, __shfl_xor(mx, m, 32));
      const float nm = fmaxf(mrow[i], mx);
      const float alpha = __expf(mrow[i] - nm);
      const float p0 = __expf(s[0][i] - nm);
      const float p1 = __expf(s[1][i] - nm);
      lrow[i] *= alpha;
      mrow[i] = nm;
#pragma unroll
      for (int j = 0; j < 4; ++j) o[j][i] *= alpha;
      pl[wave][i + hi * 8][r]      = (__bf16)p0;   // C-layout -> LDS
      pl[wave][i + hi * 8][16 + r] = (__bf16)p1;
    }
    asm volatile("s_wait_dscnt 0x0" ::: "memory");  // wave-local transpose visibility
    BV ap;                                          // reload P in A-layout
    ap.h[0] = *(const v8bf*)&pl[wave][r][hi * 8];
    ap.h[1] = *(const v8bf*)&pl[wave][r][16 + hi * 8];

    // row sums of P via WMMA against ones: every lane of a half gets its row's sum
    v8f zz = {};
    v8f ls = __builtin_amdgcn_wmma_f32_16x16x32_bf16(false, ap.v, false, onesb.v,
                                                     (short)0, zz, false, false);
#pragma unroll
    for (int i = 0; i < 8; ++i) lrow[i] += ls[i];

    // O += P @ V : B operand columns = vt rows (contiguous along keys)
#pragma unroll
    for (int j = 0; j < 4; ++j) {
      const __bf16* vp = vbase + ((size_t)(j * 16 + r) << 10) + kb + hi * 8;
      BV bv;
      bv.h[0] = *(const v8bf*)vp;
      bv.h[1] = *(const v8bf*)(vp + 16);
      o[j] = __builtin_amdgcn_wmma_f32_16x16x32_bf16(false, ap.v, false, bv.v,
                                                     (short)0, o[j], false, false);
    }
    asm volatile("s_wait_dscnt 0x0" ::: "memory");
  }

#pragma unroll
  for (int j = 0; j < 4; ++j) {
#pragma unroll
    for (int i = 0; i < 8; ++i) {
      const float val = o[j][i] * (1.0f / lrow[i]);
      out[((size_t)(b * 1024 + q0 + i + hi * 8) << 10) + h * 64 + j * 16 + r] = (__bf16)val;
    }
  }
}

// ---------------------------------------------------------------- host orchestration
extern "C" void kernel_launch(void* const* d_in, const int* in_sizes, int n_in,
                              void* d_out, int out_size, void* d_ws, size_t ws_size,
                              hipStream_t stream) {
  (void)in_sizes; (void)n_in; (void)out_size; (void)ws_size;
  const int Bz = 8, S = 1024, D = 1024, FF = 4096;
  const int M = Bz * S;  // 8192

  const float* hidden = (const float*)d_in[0];
  const float* Wq = (const float*)d_in[1];  const float* bq = (const float*)d_in[2];
  const float* Wk = (const float*)d_in[3];  const float* bk = (const float*)d_in[4];
  const float* Wv = (const float*)d_in[5];  const float* bv = (const float*)d_in[6];
  const float* Wo = (const float*)d_in[7];  const float* bo = (const float*)d_in[8];
  const float* ln1g = (const float*)d_in[9];  const float* ln1b = (const float*)d_in[10];
  const float* ln2g = (const float*)d_in[11]; const float* ln2b = (const float*)d_in[12];
  const float* W1 = (const float*)d_in[13]; const float* b1 = (const float*)d_in[14];
  const float* W2 = (const float*)d_in[15]; const float* b2 = (const float*)d_in[16];
  float* outp = (float*)d_out;

  char* ws = (char*)d_ws;
  size_t off = 0;
  auto alloc = [&](size_t bytes) -> void* {
    void* p = ws + off;
    off += (bytes + 255) & ~(size_t)255;
    return p;
  };
  __bf16* hB    = (__bf16*)alloc((size_t)M * D * 2);
  __bf16* qB    = (__bf16*)alloc((size_t)M * D * 2);
  __bf16* kB    = (__bf16*)alloc((size_t)M * D * 2);
  __bf16* vtB   = (__bf16*)alloc((size_t)M * D * 2);
  __bf16* attnB = (__bf16*)alloc((size_t)M * D * 2);
  __bf16* h2B   = (__bf16*)alloc((size_t)M * D * 2);
  __bf16* gB    = (__bf16*)alloc((size_t)M * FF * 2);
  float*  resid = (float*)alloc((size_t)M * D * 4);
  __bf16* WqB = (__bf16*)alloc((size_t)D * D * 2);
  __bf16* WkB = (__bf16*)alloc((size_t)D * D * 2);
  __bf16* WvB = (__bf16*)alloc((size_t)D * D * 2);
  __bf16* WoB = (__bf16*)alloc((size_t)D * D * 2);
  __bf16* W1B = (__bf16*)alloc((size_t)FF * D * 2);
  __bf16* W2B = (__bf16*)alloc((size_t)D * FF * 2);

  const int nDD = D * D, nFD = FF * D;
  cast_f32_bf16_k<<<(nDD + 255) / 256, 256, 0, stream>>>(Wq, WqB, nDD);
  cast_f32_bf16_k<<<(nDD + 255) / 256, 256, 0, stream>>>(Wk, WkB, nDD);
  cast_f32_bf16_k<<<(nDD + 255) / 256, 256, 0, stream>>>(Wv, WvB, nDD);
  cast_f32_bf16_k<<<(nDD + 255) / 256, 256, 0, stream>>>(Wo, WoB, nDD);
  cast_f32_bf16_k<<<(nFD + 255) / 256, 256, 0, stream>>>(W1, W1B, nFD);
  cast_f32_bf16_k<<<(nFD + 255) / 256, 256, 0, stream>>>(W2, W2B, nFD);

  ln_bf16_k<<<M, 256, 0, stream>>>(hidden, ln1g, ln1b, hB);

  dim3 gDD(D / 64, M / 128);
  gemm_bf16_k<0><<<gDD, 256, 0, stream>>>(hB, WqB, bq, nullptr, nullptr, qB,  M, D, D, 0.125f);
  gemm_bf16_k<0><<<gDD, 256, 0, stream>>>(hB, WkB, bk, nullptr, nullptr, kB,  M, D, D, 1.0f);
  gemm_bf16_k<1><<<gDD, 256, 0, stream>>>(hB, WvB, bv, nullptr, nullptr, vtB, M, D, D, 1.0f);

  attn_k<<<dim3(S / 128, Bz * 16), 256, 0, stream>>>(qB, kB, vtB, attnB);

  gemm_bf16_k<2><<<gDD, 256, 0, stream>>>(attnB, WoB, bo, hidden, resid, nullptr, M, D, D, 1.0f);

  ln_bf16_k<<<M, 256, 0, stream>>>(resid, ln2g, ln2b, h2B);

  gemm_bf16_k<3><<<dim3(FF / 64, M / 128), 256, 0, stream>>>(h2B, W1B, b1, nullptr, nullptr, gB, M, FF, D, 1.0f);
  gemm_bf16_k<2><<<dim3(D / 64, M / 128), 256, 0, stream>>>(gB, W2B, b2, resid, outp, nullptr, M, D, FF, 1.0f);
}
